// Encoder_24240795419242
// MI455X (gfx1250) — matrix operand order repaired and checked by reading, not compiled, not run on hardware
//
#include <hip/hip_runtime.h>
#include <hip/hip_bf16.h>

#define N_ROWS 262144
#define DIMM   128
#define RNUM   20
#define ENUM   32768
#define EPSV   1e-5f

typedef __attribute__((ext_vector_type(16))) __bf16 v16bf;
typedef __attribute__((ext_vector_type(8)))  float  v8f;

// ---------------- workspace layout (bytes) ----------------
// counts[R] | cursor[R] | offsets[R+1] | sum[R*DIM] | sq[R*DIM]
// | scale[R*DIM] | shift[R*DIM] | tcnt[E] | perm[N]
#define O_COUNTS  0
#define O_CURSOR  128
#define O_OFFSETS 256
#define O_SUM     512
#define O_SQ      (O_SUM   + RNUM*DIMM*4)     // 10752
#define O_SCALE   (O_SQ    + RNUM*DIMM*4)     // 20992
#define O_SHIFT   (O_SCALE + RNUM*DIMM*4)     // 31232
#define O_TCNT    (O_SHIFT + RNUM*DIMM*4)     // 41472
#define O_PERM    (O_TCNT  + ENUM*4)          // 172544
#define ZERO_WORDS (O_PERM / 4)               // zero counts..tcnt

__global__ void zero_f32(float* __restrict__ p, int n) {
    int i = blockIdx.x * 256 + threadIdx.x;
    if (i < n) p[i] = 0.0f;
}

__global__ void count_kernel(const int* __restrict__ rel,
                             const int* __restrict__ tgt,
                             int* __restrict__ counts,
                             float* __restrict__ tcnt, int n) {
    int i = blockIdx.x * 256 + threadIdx.x;
    if (i < n) {
        atomicAdd(&counts[rel[i]], 1);
        atomicAdd(&tcnt[tgt[i]], 1.0f);
    }
}

__global__ void prefix_kernel(const int* __restrict__ counts,
                              int* __restrict__ offsets) {
    if (threadIdx.x == 0) {
        int s = 0;
        for (int r = 0; r < RNUM; ++r) { offsets[r] = s; s += counts[r]; }
        offsets[RNUM] = s;
    }
}

__global__ void scatter_kernel(const int* __restrict__ rel,
                               const int* __restrict__ offsets,
                               int* __restrict__ cursor,
                               int* __restrict__ perm, int n) {
    int i = blockIdx.x * 256 + threadIdx.x;
    if (i < n) {
        int r = rel[i];
        int p = atomicAdd(&cursor[r], 1);
        perm[offsets[r] + p] = i;
    }
}

__global__ void finalize_kernel(const float* __restrict__ gsum,
                                const float* __restrict__ gsq,
                                const int* __restrict__ counts,
                                const float* __restrict__ gamma,
                                const float* __restrict__ beta,
                                float* __restrict__ scale,
                                float* __restrict__ shiftv) {
    int i = blockIdx.x * 256 + threadIdx.x;
    if (i < RNUM * DIMM) {
        int r = i >> 7;
        float c = (float)counts[r];
        float d = fmaxf(c, 1.0f);
        float mean = gsum[i] / d;
        float var = fmaxf(gsq[i] / d - mean * mean, 0.0f);
        if (c > 1.0f) {
            float s = gamma[i] * rsqrtf(var + EPSV);
            scale[i] = s;
            shiftv[i] = beta[i] - mean * s;
        } else {
            scale[i] = 1.0f;
            shiftv[i] = 0.0f;
        }
    }
}

__global__ void divide_kernel(float* __restrict__ out,
                              const float* __restrict__ tcnt) {
    int i = blockIdx.x * 256 + threadIdx.x;
    if (i < ENUM * DIMM) out[i] = out[i] / fmaxf(tcnt[i >> 7], 1.0f);
}

// Gathered per-relation GEMM: block = 256 threads (8 wave32), each wave owns a
// 16-row x 128-col output tile. W[r] staged transposed in LDS as bf16.
// STATS=true: accumulate bucket sum/sumsq.  STATS=false: BN+ReLU+scatter-add.
template<bool STATS>
__global__ __launch_bounds__(256) void trans_gemm_kernel(
    const float* __restrict__ x,
    const float* __restrict__ W,
    const float* __restrict__ bias,
    const int* __restrict__ perm,
    const int* __restrict__ offsets,
    const int* __restrict__ counts,
    const int* __restrict__ target_ids,
    const float* __restrict__ scale,
    const float* __restrict__ shiftv,
    float* __restrict__ gsum,
    float* __restrict__ gsq,
    float* __restrict__ out)
{
    __shared__ __bf16 sW[DIMM * DIMM];   // W[r] transposed: sW[n*128 + k]
    __shared__ float sSum[DIMM];
    __shared__ float sSq[DIMM];

    const int r = blockIdx.y;
    const int cnt = counts[r];
    const int rowBase = blockIdx.x * 128;
    if (rowBase >= cnt) return;          // block-uniform early exit
    const int off = offsets[r];

    // cooperative stage of W[r] (fp32 -> bf16, transposed)
    const float* Wr = W + (size_t)r * DIMM * DIMM;
    for (int idx = threadIdx.x; idx < DIMM * DIMM; idx += 256) {
        int k = idx >> 7;
        int n = idx & 127;
        sW[n * DIMM + k] = (__bf16)Wr[idx];
    }
    if (STATS) {
        if (threadIdx.x < DIMM) { sSum[threadIdx.x] = 0.0f; sSq[threadIdx.x] = 0.0f; }
    }
    __syncthreads();

    const int wave = threadIdx.x >> 5;
    const int lane = threadIdx.x & 31;
    const int nl   = lane & 15;
    const int half = lane >> 4;
    const int tileBase = rowBase + wave * 16;

    if (tileBase < cnt) {
        // A-matrix row for this lane (rows gathered through perm; clamp tail)
        int aIdx = tileBase + nl;
        if (aIdx > cnt - 1) aIdx = cnt - 1;
        const float* xrow = x + (size_t)perm[off + aIdx] * DIMM;

        v8f acc[8] = {};

        #pragma unroll
        for (int t = 0; t < 4; ++t) {                // K tiles of 32
            // A frag, 16-bit A 16x32 layout (ISA 7.12.2):
            // VGPR i<4 : K = 8*half + 2i ; VGPR i>=4 : K = 16 + 8*half + 2(i-4)
            v16bf afrag;
            #pragma unroll
            for (int i = 0; i < 8; ++i) {
                const int ka = t * 32 +
                    ((i < 4) ? (8 * half + 2 * i) : (16 + 8 * half + 2 * (i - 4)));
                const float2 v = *(const float2*)(xrow + ka);
                afrag[2 * i]     = (__bf16)v.x;
                afrag[2 * i + 1] = (__bf16)v.y;
            }
            #pragma unroll
            for (int nt = 0; nt < 8; ++nt) {         // 8 N tiles of 16
                // B frag 32x16: same K order as A (i<4: K=8*half+2i; i>=4: K=16+8*half+2(i-4))
                const __bf16* wc = sW + (nt * 16 + nl) * DIMM + t * 32;
                v16bf bfrag;
                #pragma unroll
                for (int i = 0; i < 8; ++i) {
                    const int kb = (i < 4) ? (8 * half + 2 * i) : (16 + 8 * half + 2 * (i - 4));
                    bfrag[2 * i]     = wc[kb];
                    bfrag[2 * i + 1] = wc[kb + 1];
                }
                acc[nt] = __builtin_amdgcn_wmma_f32_16x16x32_bf16(
                    false, afrag, false, bfrag, (short)0, acc[nt], false, false);
            }
        }

        if (STATS) {
            #pragma unroll
            for (int nt = 0; nt < 8; ++nt) {
                const int col = nt * 16 + nl;
                const float bb = bias[r * DIMM + col];
                float s = 0.0f, s2 = 0.0f;
                #pragma unroll
                for (int j = 0; j < 8; ++j) {        // row m = j + 8*half
                    if (tileBase + j + 8 * half < cnt) {
                        const float z = acc[nt][j] + bb;
                        s += z; s2 += z * z;
                    }
                }
                atomicAdd(&sSum[col], s);
                atomicAdd(&sSq[col], s2);
            }
        } else {
            int tg[8];
            #pragma unroll
            for (int j = 0; j < 8; ++j) {
                int ridx = tileBase + j + 8 * half;
                if (ridx > cnt - 1) ridx = cnt - 1;
                tg[j] = target_ids[perm[off + ridx]];
            }
            #pragma unroll
            for (int nt = 0; nt < 8; ++nt) {
                const int col = nt * 16 + nl;
                const float bb = bias[r * DIMM + col];
                const float sc = scale[r * DIMM + col];
                const float sh = shiftv[r * DIMM + col];
                #pragma unroll
                for (int j = 0; j < 8; ++j) {
                    if (tileBase + j + 8 * half < cnt) {
                        float z = acc[nt][j] + bb;
                        z = fmaxf(z * sc + sh, 0.0f);
                        unsafeAtomicAdd(&out[(size_t)tg[j] * DIMM + col], z);
                    }
                }
            }
        }
    }

    if (STATS) {
        __syncthreads();
        if (threadIdx.x < DIMM) {
            unsafeAtomicAdd(&gsum[r * DIMM + threadIdx.x], sSum[threadIdx.x]);
            unsafeAtomicAdd(&gsq[r * DIMM + threadIdx.x],  sSq[threadIdx.x]);
        }
    }
}

extern "C" void kernel_launch(void* const* d_in, const int* in_sizes, int n_in,
                              void* d_out, int out_size, void* d_ws, size_t ws_size,
                              hipStream_t stream) {
    const float* x     = (const float*)d_in[0];
    const float* W     = (const float*)d_in[1];
    const float* b     = (const float*)d_in[2];
    const float* gamma = (const float*)d_in[3];
    const float* beta  = (const float*)d_in[4];
    const int*   rel   = (const int*)d_in[5];
    const int*   tgt   = (const int*)d_in[6];
    float* out = (float*)d_out;

    char* ws = (char*)d_ws;
    int*   counts  = (int*)(ws + O_COUNTS);
    int*   cursor  = (int*)(ws + O_CURSOR);
    int*   offsets = (int*)(ws + O_OFFSETS);
    float* gsum    = (float*)(ws + O_SUM);
    float* gsq     = (float*)(ws + O_SQ);
    float* scale   = (float*)(ws + O_SCALE);
    float* shiftv  = (float*)(ws + O_SHIFT);
    float* tcnt    = (float*)(ws + O_TCNT);
    int*   perm    = (int*)(ws + O_PERM);

    // 1) zero scratch + output (harness poisons with 0xAA)
    zero_f32<<<(ZERO_WORDS + 255) / 256, 256, 0, stream>>>((float*)ws, ZERO_WORDS);
    zero_f32<<<(ENUM * DIMM + 255) / 256, 256, 0, stream>>>(out, ENUM * DIMM);

    // 2) per-relation row counts + per-target counts
    count_kernel<<<N_ROWS / 256, 256, 0, stream>>>(rel, tgt, counts, tcnt, N_ROWS);

    // 3) prefix sum over R=20
    prefix_kernel<<<1, 32, 0, stream>>>(counts, offsets);

    // 4) compact row indices per relation
    scatter_kernel<<<N_ROWS / 256, 256, 0, stream>>>(rel, offsets, cursor, perm, N_ROWS);

    // 5) gathered GEMM pass 1: bucket sum / sumsq (z = xW + b)
    dim3 ggrid(N_ROWS / 128, RNUM);
    trans_gemm_kernel<true><<<ggrid, 256, 0, stream>>>(
        x, W, b, perm, offsets, counts, tgt, scale, shiftv, gsum, gsq, out);

    // 6) BN scale/shift per (relation, dim)
    finalize_kernel<<<(RNUM * DIMM + 255) / 256, 256, 0, stream>>>(
        gsum, gsq, counts, gamma, beta, scale, shiftv);

    // 7) gathered GEMM pass 2: recompute z, BN + ReLU, scatter-add to targets
    trans_gemm_kernel<false><<<ggrid, 256, 0, stream>>>(
        x, W, b, perm, offsets, counts, tgt, scale, shiftv, gsum, gsq, out);

    // 8) mean pooling: divide by per-target counts
    divide_kernel<<<(ENUM * DIMM + 255) / 256, 256, 0, stream>>>(out, tcnt);
}
